// Attention_8117488189778
// MI455X (gfx1250) — matrix operand order repaired and checked by reading, not compiled, not run on hardware
//
#include <hip/hip_runtime.h>
#include <hip/hip_bf16.h>
#include <stdint.h>

// Problem constants (match reference)
constexpr int B_  = 2;
constexpr int S_  = 2048;
constexpr int H_  = 1024;
constexpr int NH_ = 16;
constexpr int HD_ = 64;           // head dim
constexpr int TOK = B_ * S_;      // 4096 tokens

// Workspace layout (bytes)
constexpr size_t SZ_X = (size_t)TOK * H_ * 2;  // bf16 activations: 8 MiB each
constexpr size_t SZ_W = (size_t)H_ * H_ * 2;   // bf16 weights: 2 MiB each
constexpr size_t OFF_XQ = 0;
constexpr size_t OFF_XK = OFF_XQ + SZ_X;
constexpr size_t OFF_XV = OFF_XK + SZ_X;
constexpr size_t OFF_WQ = OFF_XV + SZ_X;
constexpr size_t OFF_WK = OFF_WQ + SZ_W;
constexpr size_t OFF_WV = OFF_WK + SZ_W;
constexpr size_t OFF_QH = OFF_WV + SZ_W;       // [B,NH,S,HD] bf16
constexpr size_t OFF_KH = OFF_QH + SZ_X;       // [B,NH,S,HD] bf16
constexpr size_t OFF_VT = OFF_KH + SZ_X;       // [B,NH,HD,S] bf16 (transposed)

#if defined(__HIP_DEVICE_COMPILE__)
typedef __bf16 bf16_t;
typedef __bf16 v16bf  __attribute__((ext_vector_type(16)));
typedef __bf16 bf16x8 __attribute__((ext_vector_type(8)));
typedef float  v8f    __attribute__((ext_vector_type(8)));
typedef unsigned int u32x4 __attribute__((ext_vector_type(4)));
typedef int          i32x8 __attribute__((ext_vector_type(8)));
typedef int          i32x4 __attribute__((ext_vector_type(4)));
// LDS (addrspace 3) pointer types for native ds_load access
typedef bf16x8 __attribute__((address_space(3))) lds_bf16x8;

// ---------------------------------------------------------------------------
// TDM: issue a 2-D tile load (rows x 32 bf16, row-major compacted) into LDS.
// D# layout per CDNA5 ISA 8.3-8.5. Groups 2/3 zero (2-D tensor).
// 6-arg builtin form (this toolchain): (v4u, v8i, v4i, v4i, v8i, i32 cpol).
// ---------------------------------------------------------------------------
__device__ __forceinline__ void tdm_load_tile_2d(const __bf16* gtile,
                                                 uint32_t lds_off_bytes,
                                                 int tile_rows,
                                                 int tensor_rows,
                                                 int row_len_elems) {
  const uint64_t ga = (uint64_t)(uintptr_t)gtile;
  u32x4 g0;
  g0[0] = 1u;                                            // count=1, user mode
  g0[1] = lds_off_bytes;                                 // lds_addr
  g0[2] = (uint32_t)ga;                                  // global_addr[31:0]
  g0[3] = (uint32_t)((ga >> 32) & 0x1FFFFFFu) | (2u << 30);  // addr[56:32] | type=2
  i32x8 g1;
  g1[0] = 0x00010000;                                    // data_size=1 (2 bytes)
  g1[1] = (int)(((uint32_t)row_len_elems & 0xFFFFu) << 16);       // tensor_dim0 lo16
  g1[2] = (int)((((uint32_t)row_len_elems >> 16) & 0xFFFFu) |
                (((uint32_t)tensor_rows & 0xFFFFu) << 16));       // dim0 hi | dim1 lo
  g1[3] = (int)((((uint32_t)tensor_rows >> 16) & 0xFFFFu) |
                (32u << 16));                            // dim1 hi | tile_dim0=32
  g1[4] = tile_rows;                                     // tile_dim1
  g1[5] = row_len_elems;                                 // tensor_dim0_stride lo32
  g1[6] = 0;
  g1[7] = 0;
  i32x4 z4 = {0, 0, 0, 0};
  i32x8 z8 = {0, 0, 0, 0, 0, 0, 0, 0};
  __builtin_amdgcn_tensor_load_to_lds(g0, g1, z4, z4, z8, 0);
}
#endif

// ---------------------------------------------------------------------------
// fp32 -> bf16 conversion
// ---------------------------------------------------------------------------
__global__ __launch_bounds__(256) void cvt_f32_bf16(const float* __restrict__ in,
                                                    unsigned short* __restrict__ out,
                                                    int n) {
#if defined(__HIP_DEVICE_COMPILE__)
  int i = blockIdx.x * blockDim.x + threadIdx.x;
  if (i < n) {
    __bf16* o = reinterpret_cast<__bf16*>(out);
    o[i] = (__bf16)in[i];
  }
#endif
}

// ---------------------------------------------------------------------------
// QKV projection GEMM with TDM-staged LDS tiles, double buffered.
// Block = 8 waves computes a 128x64 macro tile; each wave owns 32x32 (2x2 WMMA).
// out[token, col] = sum_k X[token,k] * W[col,k] + bias[col]
// Q/K -> [B,NH,S,HD]; V (transposeV=1) -> [B,NH,HD,S].
// ---------------------------------------------------------------------------
constexpr int XTILE_ELEMS = 128 * 32;   // 8KB each (bf16)
constexpr int WTILE_ELEMS = 64 * 32;    // 4KB each (bf16)
// LDS byte offsets within the single __shared__ block (assumed base 0)
#define XTILE_OFF(buf) ((uint32_t)(buf) * 8192u)            // 128*32*2 = 8KB each
#define WTILE_OFF(buf) (16384u + (uint32_t)(buf) * 4096u)   // 64*32*2  = 4KB each

__global__ __launch_bounds__(256) void qkv_gemm(const unsigned short* __restrict__ Xus,
                                                const unsigned short* __restrict__ Wus,
                                                const float* __restrict__ bias,
                                                unsigned short* __restrict__ outus,
                                                int transposeV) {
#if defined(__HIP_DEVICE_COMPILE__)
  __shared__ __align__(128) __bf16 lds_pool[2 * XTILE_ELEMS + 2 * WTILE_ELEMS]; // 24KB

  // Keep the LDS allocation alive and mark it may-be-written (the TDM engine
  // writes it behind the compiler's back).
  {
    __bf16* escape = lds_pool;
    asm volatile("" : "+v"(escape) :: "memory");
  }

  const __bf16* X = reinterpret_cast<const __bf16*>(Xus);
  const __bf16* W = reinterpret_cast<const __bf16*>(Wus);
  __bf16* out = reinterpret_cast<__bf16*>(outus);

  const int lane = threadIdx.x & 31;
  const int wv   = threadIdx.x >> 5;     // 0..7
  const int wr   = wv >> 1;              // 0..3 (row group of 32)
  const int wc   = wv & 1;               // 0..1 (col group of 32)

  const int NTC = H_ / 64;               // 16 col macro-tiles
  const int rowStart = (blockIdx.x / NTC) * 128;
  const int colStart = (blockIdx.x % NTC) * 64;

  const int l15  = lane & 15;
  const int half = lane >> 4;
  const int kb8  = half * 8;
  const int kb16 = half * 16;

  // Prologue: wave 0 stages k=0 tiles via TDM
  if (threadIdx.x < 32) {
    tdm_load_tile_2d(X + (size_t)rowStart * H_ + 0, XTILE_OFF(0), 128, TOK, H_);
    tdm_load_tile_2d(W + (size_t)colStart * H_ + 0, WTILE_OFF(0), 64,  H_,  H_);
  }

  v8f acc[2][2] = { { v8f{}, v8f{} }, { v8f{}, v8f{} } };
  int cur = 0;

  for (int k = 0; k < H_; k += 32) {
    if (threadIdx.x < 32) __builtin_amdgcn_s_wait_tensorcnt(0);
    __syncthreads();   // cur buffer ready for everyone; prior reads of next done

    if (k + 32 < H_ && threadIdx.x < 32) {
      tdm_load_tile_2d(X + (size_t)rowStart * H_ + (k + 32), XTILE_OFF(cur ^ 1),
                       128, TOK, H_);
      tdm_load_tile_2d(W + (size_t)colStart * H_ + (k + 32), WTILE_OFF(cur ^ 1),
                       64, H_, H_);
    }

    // Opaque LDS byte offsets (launder through asm so the TDM-written tiles
    // cannot be provenance-traced to the never-visibly-stored lds_pool, and
    // keep them as addrspace(3) so the reads lower to native ds_load_b128).
    uint32_t xbase = XTILE_OFF(cur);
    uint32_t wbase = WTILE_OFF(cur);
    asm volatile("" : "+v"(xbase), "+v"(wbase));

    // A fragments (16x32) for the two 16-row sub-tiles this wave owns
    v16bf af[2];
#pragma unroll
    for (int mi = 0; mi < 2; ++mi) {
      const int row = wr * 32 + mi * 16 + l15;
      const lds_bf16x8* ap =
          (const lds_bf16x8*)(xbase + (uint32_t)(row * 64 + kb8 * 2));
      bf16x8 a0 = ap[0];   // bytes +0  : K = kb8 .. kb8+7
      bf16x8 a1 = ap[2];   // bytes +32 : K = 16+kb8 .. 16+kb8+7
      af[mi] = __builtin_shufflevector(a0, a1, 0,1,2,3,4,5,6,7,8,9,10,11,12,13,14,15);
    }
    // B fragments (32x16) for the two 16-col sub-tiles
    v16bf bfr[2];
#pragma unroll
    for (int ni = 0; ni < 2; ++ni) {
      const int col = wc * 32 + ni * 16 + l15;
      const lds_bf16x8* bp =
          (const lds_bf16x8*)(wbase + (uint32_t)(col * 64 + kb8 * 2));
      bf16x8 b0 = bp[0];   // K = kb8 .. kb8+7
      bf16x8 b1 = bp[2];   // K = 16+kb8 .. 16+kb8+7
      bfr[ni] = __builtin_shufflevector(b0, b1, 0,1,2,3,4,5,6,7,8,9,10,11,12,13,14,15);
    }
#pragma unroll
    for (int mi = 0; mi < 2; ++mi)
#pragma unroll
      for (int ni = 0; ni < 2; ++ni)
        acc[mi][ni] = __builtin_amdgcn_wmma_f32_16x16x32_bf16(
            false, af[mi], false, bfr[ni], (short)0, acc[mi][ni], false, false);

    cur ^= 1;
  }

  // Epilogue: bias + store bf16 in head layout
#pragma unroll
  for (int ni = 0; ni < 2; ++ni) {
    const int col = colStart + wc * 32 + ni * 16 + l15;
    const float bval = bias[col];
    const int h = col / HD_;
    const int d = col % HD_;
#pragma unroll
    for (int mi = 0; mi < 2; ++mi) {
#pragma unroll
      for (int r = 0; r < 8; ++r) {
        const int token = rowStart + wr * 32 + mi * 16 + r + 8 * half;
        const int b = token / S_;
        const int s = token % S_;
        const __bf16 v = (__bf16)(acc[mi][ni][r] + bval);
        if (transposeV == 0)
          out[((size_t)(b * NH_ + h) * S_ + s) * HD_ + d] = v;
        else
          out[((size_t)(b * NH_ + h) * HD_ + d) * S_ + s] = v;
      }
    }
  }
#endif
}

// ---------------------------------------------------------------------------
// Flash attention: one wave per 16-query tile of one (b,h). Streams keys in
// chunks of 32 with online softmax; ctx accumulated in f32 via WMMA against
// transposed V. Writes fp32 [B,S,H].
// ---------------------------------------------------------------------------
__global__ __launch_bounds__(256) void flash_attn(const unsigned short* __restrict__ Qus,
                                                  const unsigned short* __restrict__ Kus,
                                                  const unsigned short* __restrict__ Vus,
                                                  const float* __restrict__ mask,
                                                  float* __restrict__ out) {
#if defined(__HIP_DEVICE_COMPILE__)
  __shared__ __align__(32) __bf16 pbuf[8][16 * 32];   // per-wave P tile

  const __bf16* Qh = reinterpret_cast<const __bf16*>(Qus);
  const __bf16* Kh = reinterpret_cast<const __bf16*>(Kus);
  const __bf16* Vt = reinterpret_cast<const __bf16*>(Vus);

  const int lane  = threadIdx.x & 31;
  const int wslot = threadIdx.x >> 5;
  const int wid   = (blockIdx.x * blockDim.x + threadIdx.x) >> 5;
  const int QT    = S_ / 16;                // 128 q tiles per head
  const int qt    = wid % QT;
  const int bh    = wid / QT;               // 0..B*NH-1
  const int b     = bh / NH_;
  const int h     = bh % NH_;

  const __bf16* Q = Qh + (size_t)bh * S_ * HD_;
  const __bf16* K = Kh + (size_t)bh * S_ * HD_;
  const __bf16* V = Vt + (size_t)bh * HD_ * S_;   // [HD][S]
  const float* mrow = mask + (size_t)b * S_;      // [B,1,1,S] additive

  const int l15  = lane & 15;
  const int half = lane >> 4;
  const int kb8  = half * 8;
  const int kb16 = half * 16;

  // Q A-fragments for d-chunks {0..31}, {32..63}
  v16bf qa[2];
  const int qrow = qt * 16 + l15;
#pragma unroll
  for (int t = 0; t < 2; ++t) {
    bf16x8 a0 = *(const bf16x8*)(Q + qrow * HD_ + t * 32 + kb8);
    bf16x8 a1 = *(const bf16x8*)(Q + qrow * HD_ + t * 32 + 16 + kb8);
    qa[t] = __builtin_shufflevector(a0, a1, 0,1,2,3,4,5,6,7,8,9,10,11,12,13,14,15);
  }

  float rm[8], rl[8];
  v8f acc[4] = {v8f{}, v8f{}, v8f{}, v8f{}};
#pragma unroll
  for (int r = 0; r < 8; ++r) { rm[r] = -1e30f; rl[r] = 0.0f; }
  const float sm_scale = 0.125f;   // 1/sqrt(64)

  __bf16* pt = pbuf[wslot];

  for (int j = 0; j < S_; j += 32) {
    // ---- scores: two 16-key tiles, K-dim = HD = 64 (2 WMMAs each) ----
    v8f sc[2];
#pragma unroll
    for (int t = 0; t < 2; ++t) {
      const int keyN = j + t * 16 + l15;
      v8f c = {};
#pragma unroll
      for (int dc = 0; dc < 2; ++dc) {
        // B in table order: K[keyN][dc*32 + kb8 + e], K[keyN][dc*32 + 16 + kb8 + e]
        bf16x8 k0 = *(const bf16x8*)(K + keyN * HD_ + dc * 32 + kb8);
        bf16x8 k1 = *(const bf16x8*)(K + keyN * HD_ + dc * 32 + 16 + kb8);
        v16bf kb = __builtin_shufflevector(k0, k1, 0,1,2,3,4,5,6,7,8,9,10,11,12,13,14,15);
        c = __builtin_amdgcn_wmma_f32_16x16x32_bf16(false, qa[dc], false, kb,
                                                    (short)0, c, false, false);
      }
      const float mv = mrow[keyN];
#pragma unroll
      for (int r = 0; r < 8; ++r) sc[t][r] = c[r] * sm_scale + mv;
    }

    // ---- online softmax (rows live per-VGPR; cols across 16 lanes) ----
    float corr[8];
#pragma unroll
    for (int r = 0; r < 8; ++r) {
      float mx = fmaxf(sc[0][r], sc[1][r]);
#pragma unroll
      for (int d = 1; d < 16; d <<= 1) mx = fmaxf(mx, __shfl_xor(mx, d, 16));
      const float mn = fmaxf(rm[r], mx);
      corr[r] = __expf(rm[r] - mn);
      rm[r] = mn;
      const float p0 = __expf(sc[0][r] - mn);
      const float p1 = __expf(sc[1][r] - mn);
      sc[0][r] = p0;
      sc[1][r] = p1;
      float ps = p0 + p1;
#pragma unroll
      for (int d = 1; d < 16; d <<= 1) ps += __shfl_xor(ps, d, 16);
      rl[r] = rl[r] * corr[r] + ps;
    }
#pragma unroll
    for (int t = 0; t < 4; ++t)
#pragma unroll
      for (int r = 0; r < 8; ++r) acc[t][r] *= corr[r];

    // ---- re-layout P (D-layout f32 -> A-layout bf16) via wave-private LDS ----
#pragma unroll
    for (int t = 0; t < 2; ++t) {
      const int col = t * 16 + l15;
#pragma unroll
      for (int r = 0; r < 8; ++r) {
        const int rr = r + 8 * half;
        pt[rr * 32 + col] = (__bf16)sc[t][r];
      }
    }
    asm volatile("s_wait_dscnt 0" ::: "memory");   // DS ops are in-order per wave
    v16bf pa;
    {
      bf16x8 a0 = *(const bf16x8*)(pt + l15 * 32 + kb8);
      bf16x8 a1 = *(const bf16x8*)(pt + l15 * 32 + 16 + kb8);
      pa = __builtin_shufflevector(a0, a1, 0,1,2,3,4,5,6,7,8,9,10,11,12,13,14,15);
    }

    // ---- ctx accumulate: acc[t] += P(16x32) x V(32x16) for 4 d-tiles ----
#pragma unroll
    for (int t = 0; t < 4; ++t) {
      const int dN = t * 16 + l15;
      // B in table order: Vt[dN][j + kb8 + e], Vt[dN][j + 16 + kb8 + e]
      bf16x8 v0 = *(const bf16x8*)(V + (size_t)dN * S_ + j + kb8);
      bf16x8 v1 = *(const bf16x8*)(V + (size_t)dN * S_ + j + 16 + kb8);
      v16bf vb = __builtin_shufflevector(v0, v1, 0,1,2,3,4,5,6,7,8,9,10,11,12,13,14,15);
      acc[t] = __builtin_amdgcn_wmma_f32_16x16x32_bf16(false, pa, false, vb,
                                                       (short)0, acc[t], false, false);
    }
  }

  // ---- epilogue: normalize and write fp32 [B,S,H] ----
#pragma unroll
  for (int t = 0; t < 4; ++t) {
#pragma unroll
    for (int r = 0; r < 8; ++r) {
      const int srow = qt * 16 + r + 8 * half;
      const int d    = t * 16 + l15;
      out[((size_t)b * S_ + srow) * H_ + h * HD_ + d] = acc[t][r] / rl[r];
    }
  }
#endif
}

// ---------------------------------------------------------------------------
// Host launch
// ---------------------------------------------------------------------------
extern "C" void kernel_launch(void* const* d_in, const int* in_sizes, int n_in,
                              void* d_out, int out_size, void* d_ws, size_t ws_size,
                              hipStream_t stream) {
  const float* q_in  = (const float*)d_in[0];
  const float* k_in  = (const float*)d_in[1];
  const float* v_in  = (const float*)d_in[2];
  const float* amask = (const float*)d_in[3];
  const float* Wq    = (const float*)d_in[4];
  const float* bq    = (const float*)d_in[5];
  const float* Wk    = (const float*)d_in[6];
  const float* bk    = (const float*)d_in[7];
  const float* Wv    = (const float*)d_in[8];
  const float* bv    = (const float*)d_in[9];
  float* out = (float*)d_out;

  char* ws = (char*)d_ws;
  unsigned short* Xq = (unsigned short*)(ws + OFF_XQ);
  unsigned short* Xk = (unsigned short*)(ws + OFF_XK);
  unsigned short* Xv = (unsigned short*)(ws + OFF_XV);
  unsigned short* Wqb = (unsigned short*)(ws + OFF_WQ);
  unsigned short* Wkb = (unsigned short*)(ws + OFF_WK);
  unsigned short* Wvb = (unsigned short*)(ws + OFF_WV);
  unsigned short* Qh = (unsigned short*)(ws + OFF_QH);
  unsigned short* Kh = (unsigned short*)(ws + OFF_KH);
  unsigned short* Vt = (unsigned short*)(ws + OFF_VT);

  const int nX = TOK * H_;   // 4,194,304
  const int nW = H_ * H_;    // 1,048,576

  cvt_f32_bf16<<<nX / 256, 256, 0, stream>>>(q_in, Xq, nX);
  cvt_f32_bf16<<<nX / 256, 256, 0, stream>>>(k_in, Xk, nX);
  cvt_f32_bf16<<<nX / 256, 256, 0, stream>>>(v_in, Xv, nX);
  cvt_f32_bf16<<<nW / 256, 256, 0, stream>>>(Wq, Wqb, nW);
  cvt_f32_bf16<<<nW / 256, 256, 0, stream>>>(Wk, Wkb, nW);
  cvt_f32_bf16<<<nW / 256, 256, 0, stream>>>(Wv, Wvb, nW);

  // 128x64 macro tiles: (4096/128) * (1024/64) = 32*16 = 512 blocks
  const int gemm_blocks = (TOK / 128) * (H_ / 64);
  qkv_gemm<<<gemm_blocks, 256, 0, stream>>>(Xq, Wqb, bq, Qh, 0);
  qkv_gemm<<<gemm_blocks, 256, 0, stream>>>(Xk, Wkb, bk, Kh, 0);
  qkv_gemm<<<gemm_blocks, 256, 0, stream>>>(Xv, Wvb, bv, Vt, 1);

  // B*NH*(S/16) = 4096 q-tiles, 8 waves per block -> 512 blocks
  const int attn_blocks = B_ * NH_ * (S_ / 16) / 8;
  flash_attn<<<attn_blocks, 256, 0, stream>>>(Qh, Kh, Vt, amask, out);
}
